// MixMOE_59055800320700
// MI455X (gfx1250) — compile-verified, matrix operand order checked by reading, not run on hardware
//
#include <hip/hip_runtime.h>
#include <hip/hip_bf16.h>
#include <math.h>

// MixMOE forward on gfx1250: routed top-2 MoE with bf16 WMMA expert GEMMs.
// T=2048 tokens, D=1024 d_model, H=2048 ffn hidden, E=16 experts, top-2.

#define TOK_T 2048
#define DIM_D 1024
#define DIM_H 2048
#define NEXP  16

#define TM 64          // M tile (tokens)
#define TN 64          // N tile (output cols)
#define TK 32          // K step (bf16 WMMA K)
#define LDT 36         // A-tile LDS row stride in ushorts (72B rows: 8B-aligned, conflict-free)

typedef __attribute__((ext_vector_type(16))) __bf16        v16bf;
typedef __attribute__((ext_vector_type(2)))  __bf16        v2bf;
typedef __attribute__((ext_vector_type(2)))  float         v2f;
typedef __attribute__((ext_vector_type(8)))  float         v8f;
typedef __attribute__((ext_vector_type(8)))  unsigned int  v8u;

// fp32 -> bf16 (RNE, native fptrunc)
__device__ __forceinline__ unsigned short bf16bits(float f) {
    return __builtin_bit_cast(unsigned short, (__bf16)f);
}

// pack two fp32 into one dword of bf16 (lo = a, hi = b) -> single v_cvt_pk_bf16_f32
__device__ __forceinline__ unsigned int pack2bf(float a, float b) {
    v2f f; f.x = a; f.y = b;
    v2bf p = __builtin_convertvector(f, v2bf);
    return __builtin_bit_cast(unsigned int, p);
}

__device__ __forceinline__ float gelu_tanh(float v) {
    const float c = 0.7978845608028654f;
    float t = tanhf(c * (v + 0.044715f * v * v * v));
    return 0.5f * v * (1.0f + t);
}

// A fragment: 16x32 bf16 (CDNA5 ISA 7.12.2 16-bit A layout) from row-major
// LDS tile [row][k], row stride LDT ushorts.
// lane l: row = row_base + (l&15); dword v holds K = kk, kk+1 with
// kk = (v>>2)*16 + (l>>4)*8 + (v&3)*2
__device__ __forceinline__ v16bf load_fragA(const unsigned short* lds, int row_base, int lane) {
    int r = lane & 15;
    int g = lane >> 4;
    v8u u;
#pragma unroll
    for (int v = 0; v < 8; ++v) {
        int kk = ((v >> 2) << 4) + (g << 3) + ((v & 3) << 1);
        u[v] = *(const unsigned int*)(&lds[(row_base + r) * LDT + kk]);
    }
    return __builtin_bit_cast(v16bf, u);
}

// B fragment from packed K-pair LDS layout: bs[kp * TN + n] = (bf16(2kp), bf16(2kp+1))
__device__ __forceinline__ v16bf load_fragB(const unsigned int* bs, int n_base, int lane) {
    int r = lane & 15;
    int g = lane >> 4;
    v8u u;
#pragma unroll
    for (int v = 0; v < 8; ++v) {
        int kk = ((v >> 2) << 4) + (g << 3) + ((v & 3) << 1);
        u[v] = bs[(kk >> 1) * TN + n_base + r];
    }
    return __builtin_bit_cast(v16bf, u);
}

// ---------------------------------------------------------------------------
// Kernel 1: gating. 8 tokens per block, 128 threads (8 tokens x 16 experts).
// ---------------------------------------------------------------------------
__global__ __launch_bounds__(128) void moe_gate_kernel(
    const float* __restrict__ x, const float* __restrict__ gw,
    int* __restrict__ cnt, int* __restrict__ list, float* __restrict__ wlist)
{
    __shared__ float xs[8][DIM_D];   // 32 KB
    __shared__ float lg[8][NEXP];

    int t0 = blockIdx.x * 8;
    int tid = threadIdx.x;

    for (int idx = tid; idx < 8 * DIM_D / 4; idx += 128) {
        int r = idx >> 8;             // / (DIM_D/4)
        int c = (idx & 255) * 4;
        *(float4*)&xs[r][c] = *(const float4*)&x[(size_t)(t0 + r) * DIM_D + c];
    }
    __syncthreads();

    int r = tid >> 4;
    int e = tid & 15;
    float s = 0.0f;
    for (int c = 0; c < DIM_D; ++c) s += xs[r][c] * gw[c * NEXP + e];
    lg[r][e] = s;
    __syncthreads();

    if (tid < 8) {
        int t = t0 + tid;
        float best = -INFINITY; int bi = 0;
        for (int j = 0; j < NEXP; ++j) { float v = lg[tid][j]; if (v > best) { best = v; bi = j; } }
        float best2 = -INFINITY; int bi2 = 0;
        for (int j = 0; j < NEXP; ++j) { if (j == bi) continue; float v = lg[tid][j]; if (v > best2) { best2 = v; bi2 = j; } }
        float e1 = expf(best2 - best);
        float inv = 1.0f / (1.0f + e1);
        float w0 = inv, w1 = e1 * inv;
        int p0 = atomicAdd(&cnt[bi], 1);
        list[bi * TOK_T + p0] = t;  wlist[bi * TOK_T + p0] = w0;
        int p1 = atomicAdd(&cnt[bi2], 1);
        list[bi2 * TOK_T + p1] = t; wlist[bi2 * TOK_T + p1] = w1;
    }
}

// ---------------------------------------------------------------------------
// Kernel 2: tiny exclusive scan over 16 expert counts -> h-buffer offsets.
// ---------------------------------------------------------------------------
__global__ void moe_scan_kernel(const int* __restrict__ cnt, int* __restrict__ off) {
    if (threadIdx.x == 0 && blockIdx.x == 0) {
        int s = 0;
#pragma unroll
        for (int e = 0; e < NEXP; ++e) { off[e] = s; s += cnt[e]; }
    }
}

// ---------------------------------------------------------------------------
// Kernel 3: FFN layer 1.  h[row] = gelu(x[token] @ W1[e] + b1[e])  (bf16 out)
// 64x64 block tile, K-step 32, 8 waves each with two 16x16 WMMA accumulators.
// Out-of-range rows are clamped (not masked): they only pollute C rows that
// the epilogue never stores.
// ---------------------------------------------------------------------------
__global__ __launch_bounds__(256) void moe_ffn1_kernel(
    const float* __restrict__ x, const float* __restrict__ W1, const float* __restrict__ b1,
    const int* __restrict__ cnt, const int* __restrict__ off,
    const int* __restrict__ list, unsigned short* __restrict__ hbuf)
{
    int e  = blockIdx.z;
    int m0 = blockIdx.y * TM;
    int n0 = blockIdx.x * TN;
    int ne = cnt[e];
    if (m0 >= ne) return;
    int base = off[e];

    __shared__ unsigned short As[TM * LDT];        // 4.5 KB, row-major [m][k]
    __shared__ unsigned int   Bs32[(TK / 2) * TN]; // 4 KB, packed K-pairs [k/2][n]

    int tid = threadIdx.x;

    // A staging map: 2 rows x 4 consecutive k per thread (b128 loads, b64 LDS stores)
    int ra = tid >> 3;            // 0..31
    int rb = ra + 32;
    int c4 = (tid & 7) * 4;
    int sa = m0 + ra; if (sa > ne - 1) sa = ne - 1;
    int sb = m0 + rb; if (sb > ne - 1) sb = ne - 1;
    const float* xa = x + (size_t)list[e * TOK_T + sa] * DIM_D + c4;
    const float* xb = x + (size_t)list[e * TOK_T + sb] * DIM_D + c4;

    // B staging map: 2x(2k x 2n) micro-tiles per thread
    int kp0 = tid >> 5;           // 0..7   (second tile: +8)
    int np  = (tid & 31) * 2;

    int lane = tid & 31;
    int wv = tid >> 5;
    int wm = wv & 3;
    int wn = wv >> 2;

    v8f acc0 = {};
    v8f acc1 = {};

    const float* W1e = W1 + (size_t)e * DIM_D * DIM_H;

    for (int k0 = 0; k0 < DIM_D; k0 += TK) {
        // ---- stage A (gathered x rows, fp32 -> bf16, vectorized) ----
        float4 fa = *(const float4*)(xa + k0);
        float4 fb = *(const float4*)(xb + k0);
        uint2 pa, pb;
        pa.x = pack2bf(fa.x, fa.y);
        pa.y = pack2bf(fa.z, fa.w);
        pb.x = pack2bf(fb.x, fb.y);
        pb.y = pack2bf(fb.z, fb.w);
        *(uint2*)&As[ra * LDT + c4] = pa;
        *(uint2*)&As[rb * LDT + c4] = pb;

        // ---- stage B (W1 slice -> packed K-pair bf16) ----
#pragma unroll
        for (int i = 0; i < 2; ++i) {
            int kp = kp0 + i * 8;
            const float* Wp = W1e + (size_t)(k0 + 2 * kp) * DIM_H + n0 + np;
            float2 f0 = *(const float2*)Wp;             // row k
            float2 f1 = *(const float2*)(Wp + DIM_H);   // row k+1
            uint2 w;
            w.x = pack2bf(f0.x, f1.x);                  // col np : (k, k+1)
            w.y = pack2bf(f0.y, f1.y);                  // col np+1
            *(uint2*)&Bs32[kp * TN + np] = w;
            if (k0 + TK < DIM_D)
                __builtin_prefetch(Wp + (size_t)TK * DIM_H, 0, 0);
        }
        __syncthreads();

        v16bf af  = load_fragA(As, wm * 16, lane);
        v16bf bf0 = load_fragB(Bs32, wn * 32, lane);
        v16bf bf1 = load_fragB(Bs32, wn * 32 + 16, lane);
        acc0 = __builtin_amdgcn_wmma_f32_16x16x32_bf16(false, af, false, bf0, (short)0, acc0, false, false);
        acc1 = __builtin_amdgcn_wmma_f32_16x16x32_bf16(false, af, false, bf1, (short)0, acc1, false, false);
        __syncthreads();
    }

    // epilogue: C layout -> lane l, dword i: row = i + 8*(l>>4), col = l&15
    int g  = lane >> 4;
    int nl = lane & 15;
#pragma unroll
    for (int i = 0; i < 8; ++i) {
        int ml = (g << 3) + i;
        int slot = m0 + wm * 16 + ml;
        if (slot < ne) {
            size_t row = (size_t)(base + slot);
            int c0 = n0 + wn * 32 + nl;
            float v0 = acc0[i] + b1[e * DIM_H + c0];
            hbuf[row * DIM_H + c0] = bf16bits(gelu_tanh(v0));
            int c1 = c0 + 16;
            float v1 = acc1[i] + b1[e * DIM_H + c1];
            hbuf[row * DIM_H + c1] = bf16bits(gelu_tanh(v1));
        }
    }
}

// ---------------------------------------------------------------------------
// Kernel 4: FFN layer 2 + gate-weighted scatter-combine.
// y = h @ W2[e] + b2[e];  out[token] += w * y   (fp32 atomics)
// ---------------------------------------------------------------------------
__global__ __launch_bounds__(256) void moe_ffn2_kernel(
    const unsigned short* __restrict__ hbuf, const float* __restrict__ W2,
    const float* __restrict__ b2,
    const int* __restrict__ cnt, const int* __restrict__ off,
    const int* __restrict__ list, const float* __restrict__ wlist,
    float* __restrict__ out)
{
    int e  = blockIdx.z;
    int m0 = blockIdx.y * TM;
    int n0 = blockIdx.x * TN;
    int ne = cnt[e];
    if (m0 >= ne) return;
    int base = off[e];

    __shared__ unsigned short As[TM * LDT];
    __shared__ unsigned int   Bs32[(TK / 2) * TN];

    int tid = threadIdx.x;

    // A staging: rows are compact per expert and already bf16 -> pure b64 copy
    int ra = tid >> 3;
    int rb = ra + 32;
    int c4 = (tid & 7) * 4;
    int sa = m0 + ra; if (sa > ne - 1) sa = ne - 1;
    int sb = m0 + rb; if (sb > ne - 1) sb = ne - 1;
    const unsigned short* ha = hbuf + (size_t)(base + sa) * DIM_H + c4;
    const unsigned short* hb = hbuf + (size_t)(base + sb) * DIM_H + c4;

    int kp0 = tid >> 5;
    int np  = (tid & 31) * 2;

    int lane = tid & 31;
    int wv = tid >> 5;
    int wm = wv & 3;
    int wn = wv >> 2;

    v8f acc0 = {};
    v8f acc1 = {};

    const float* W2e = W2 + (size_t)e * DIM_H * DIM_D;

    for (int k0 = 0; k0 < DIM_H; k0 += TK) {
        *(uint2*)&As[ra * LDT + c4] = *(const uint2*)(ha + k0);
        *(uint2*)&As[rb * LDT + c4] = *(const uint2*)(hb + k0);

#pragma unroll
        for (int i = 0; i < 2; ++i) {
            int kp = kp0 + i * 8;
            const float* Wp = W2e + (size_t)(k0 + 2 * kp) * DIM_D + n0 + np;
            float2 f0 = *(const float2*)Wp;
            float2 f1 = *(const float2*)(Wp + DIM_D);
            uint2 w;
            w.x = pack2bf(f0.x, f1.x);
            w.y = pack2bf(f0.y, f1.y);
            *(uint2*)&Bs32[kp * TN + np] = w;
            if (k0 + TK < DIM_H)
                __builtin_prefetch(Wp + (size_t)TK * DIM_D, 0, 0);
        }
        __syncthreads();

        v16bf af  = load_fragA(As, wm * 16, lane);
        v16bf bf0 = load_fragB(Bs32, wn * 32, lane);
        v16bf bf1 = load_fragB(Bs32, wn * 32 + 16, lane);
        acc0 = __builtin_amdgcn_wmma_f32_16x16x32_bf16(false, af, false, bf0, (short)0, acc0, false, false);
        acc1 = __builtin_amdgcn_wmma_f32_16x16x32_bf16(false, af, false, bf1, (short)0, acc1, false, false);
        __syncthreads();
    }

    int g  = lane >> 4;
    int nl = lane & 15;
#pragma unroll
    for (int i = 0; i < 8; ++i) {
        int ml = (g << 3) + i;
        int slot = m0 + wm * 16 + ml;
        if (slot < ne) {
            int   t = list[e * TOK_T + slot];
            float w = wlist[e * TOK_T + slot];
            int c0 = n0 + wn * 32 + nl;
            float v0 = acc0[i] + b2[e * DIM_D + c0];
            atomicAdd(&out[(size_t)t * DIM_D + c0], w * v0);
            int c1 = c0 + 16;
            float v1 = acc1[i] + b2[e * DIM_D + c1];
            atomicAdd(&out[(size_t)t * DIM_D + c1], w * v1);
        }
    }
}

// ---------------------------------------------------------------------------
// Host-side launch. Workspace layout:
//   [int cnt[16]] [int off[16]] [int list[E*T]] [float wlist[E*T]]
//   [ushort hbuf[2*T*H]]  (~17 MB total)
// ---------------------------------------------------------------------------
extern "C" void kernel_launch(void* const* d_in, const int* in_sizes, int n_in,
                              void* d_out, int out_size, void* d_ws, size_t ws_size,
                              hipStream_t stream) {
    const float* x   = (const float*)d_in[0];
    const float* gw  = (const float*)d_in[1];
    const float* W1  = (const float*)d_in[2];
    const float* b1  = (const float*)d_in[3];
    const float* W2  = (const float*)d_in[4];
    const float* b2  = (const float*)d_in[5];
    float* out = (float*)d_out;

    int*   cnt   = (int*)d_ws;
    int*   off   = cnt + NEXP;
    int*   list  = off + NEXP;
    float* wlist = (float*)(list + NEXP * TOK_T);
    unsigned short* hbuf = (unsigned short*)(wlist + NEXP * TOK_T);

    hipMemsetAsync(cnt, 0, NEXP * sizeof(int), stream);
    hipMemsetAsync(out, 0, (size_t)out_size * sizeof(float), stream);

    moe_gate_kernel<<<TOK_T / 8, 128, 0, stream>>>(x, gw, cnt, list, wlist);
    moe_scan_kernel<<<1, 32, 0, stream>>>(cnt, off);

    {
        dim3 grid(DIM_H / TN, TOK_T / TM, NEXP);
        moe_ffn1_kernel<<<grid, 256, 0, stream>>>(x, W1, b1, cnt, off, list, hbuf);
    }
    {
        dim3 grid(DIM_D / TN, TOK_T / TM, NEXP);
        moe_ffn2_kernel<<<grid, 256, 0, stream>>>(hbuf, W2, b2, cnt, off, list, wlist, out);
    }
}
